// FGRU_34119220199998
// MI455X (gfx1250) — matrix operand order repaired and checked by reading, not compiled, not run on hardware
//
#include <hip/hip_runtime.h>
#include <hip/hip_bf16.h>
#include <stdint.h>

// Problem constants
#define BB 64
#define TT 1024
#define DD 512
#define UU 512
#define CC 1024          // D + U
#define EPSLN 1e-3f

// Launch shape
#define NBLK 8           // persistent blocks
#define TPB 256          // 8 waves of 32
#define NWAVE 8
#define KCHUNK 256       // K staged per LDS chunk
#define APITCH (KCHUNK + 8)  // bf16 elements per staged row (pad breaks bank conflicts)

typedef __attribute__((ext_vector_type(16))) __bf16 v16bf;
typedef __attribute__((ext_vector_type(8)))  __bf16 v8bf;
typedef __attribute__((ext_vector_type(8)))  float  v8f;

__device__ __forceinline__ v16bf make_frag(v8bf lo, v8bf hi) {
  return __builtin_shufflevector(lo, hi, 0,1,2,3,4,5,6,7,8,9,10,11,12,13,14,15);
}

// ---------------------------------------------------------------------------
// Prep: transpose+convert weights to bf16 [N][K] so a WMMA B-fragment is a
// pair of 16B runs, 32B apart, per lane; convert h0 to bf16; reset barrier control.
// ---------------------------------------------------------------------------
__global__ void fgru_prep_weights(const float* __restrict__ Wg,
                                  const float* __restrict__ Wf,
                                  __bf16* __restrict__ WgT,
                                  __bf16* __restrict__ WfT) {
  const int total = CC * CC + UU * CC;
  for (int i = blockIdx.x * blockDim.x + threadIdx.x; i < total;
       i += gridDim.x * blockDim.x) {
    if (i < CC * CC) {
      int n = i / CC, k = i % CC;
      WgT[i] = (__bf16)Wg[k * CC + n];          // Wg is [C][C] row-major
    } else {
      int j = i - CC * CC;
      int n = j / CC, k = j % CC;
      WfT[j] = (__bf16)Wf[k * UU + n];          // Wf is [C][U] row-major
    }
  }
}

__global__ void fgru_prep_state(const float* __restrict__ h0,
                                __bf16* __restrict__ hb,
                                unsigned* __restrict__ ctrl) {
  int i = blockIdx.x * blockDim.x + threadIdx.x;
  if (i < BB * UU) hb[i] = (__bf16)h0[i];
  if (i < 2) ctrl[i] = 0u;
}

// ---------------------------------------------------------------------------
// Device-wide sense barrier over NBLK blocks (L2 atomics + s_sleep spin).
// ---------------------------------------------------------------------------
__device__ __forceinline__ void gbar(unsigned* cnt, unsigned* gen) {
  __threadfence();            // make a2 / hb / y stores visible device-wide
  __syncthreads();
  if (threadIdx.x == 0) {
    unsigned g = __hip_atomic_load(gen, __ATOMIC_RELAXED, __HIP_MEMORY_SCOPE_AGENT);
    unsigned a = __hip_atomic_fetch_add(cnt, 1u, __ATOMIC_ACQ_REL, __HIP_MEMORY_SCOPE_AGENT);
    if (a == (unsigned)(NBLK - 1)) {
      __hip_atomic_store(cnt, 0u, __ATOMIC_RELAXED, __HIP_MEMORY_SCOPE_AGENT);
      __hip_atomic_fetch_add(gen, 1u, __ATOMIC_RELEASE, __HIP_MEMORY_SCOPE_AGENT);
    } else {
      while (__hip_atomic_load(gen, __ATOMIC_ACQUIRE, __HIP_MEMORY_SCOPE_AGENT) == g)
        __builtin_amdgcn_s_sleep(2);
    }
  }
  __syncthreads();
}

// ---------------------------------------------------------------------------
// Persistent FGRU scan kernel.
//  Phase A: gate = sigmoid(combined @ Wg + bg); a2 = combined*gate  (bf16)
//           64 waves, wave W owns gate N-tile W (16 cols), all 4 M-tiles.
//  Phase B: z = silu(a2 @ Wf + bf); LayerNorm; write y (f32) and h (bf16).
//           blocks 0..3, block b owns rows 16b..16b+15 -> LN is intra-block.
// ---------------------------------------------------------------------------
__global__ __launch_bounds__(TPB, 1) void fgru_persistent(
    const float* __restrict__ x,      // [B][T][D] f32
    const float* __restrict__ bg,     // [C]
    const float* __restrict__ bfv,    // [U]
    const float* __restrict__ gamma,  // [U]
    const float* __restrict__ beta,   // [U]
    const __bf16* __restrict__ WgT,   // [C][C]  (n-major)
    const __bf16* __restrict__ WfT,   // [U][C]  (n-major)
    __bf16* __restrict__ hb,          // [B][U]  recurrent state
    __bf16* __restrict__ a2,          // [B][C]  gated activations
    unsigned* __restrict__ ctrl,      // barrier {count, generation}
    float* __restrict__ out)          // [B][T][U]
{
  __shared__ __align__(16) unsigned char smem[44032];
  __bf16* stageA = (__bf16*)smem;                      // [64][APITCH] bf16 (phase A)
  __bf16* stageB = (__bf16*)smem;                      // [16][APITCH] bf16 (phase B)
  float*  zbuf   = (float*)(smem + 16 * APITCH * 2 + 256); // [16][512] f32 (phase B)

  const int tid  = threadIdx.x;
  const int lane = tid & 31;
  const int wave = tid >> 5;
  const int hi   = lane >> 4;      // which half of the wave
  const int nl   = lane & 15;
  const int blk  = blockIdx.x;
  unsigned* cnt = ctrl;
  unsigned* gen = ctrl + 1;

  for (int t = 0; t < TT; ++t) {
    // ======================= Phase A: gate GEMM =======================
    const int ntileA = blk * NWAVE + wave;   // 0..63
    const int ncolA  = ntileA * 16;
    v8f acc[4] = {};                         // 4 M-tiles x 16x16 f32

    for (int kc = 0; kc < CC / KCHUNK; ++kc) {
      const int kbase = kc * KCHUNK;
      __syncthreads();
      // Stage combined[:, kbase..kbase+256) into LDS as bf16.
      if (kbase < DD) {                       // from x (f32 -> bf16)
        for (int i = tid; i < BB * KCHUNK / 2; i += TPB) {
          int kk = (i * 2) & (KCHUNK - 1);
          int m  = (i * 2) >> 8;
          float2 v = *(const float2*)&x[((size_t)m * TT + t) * DD + kbase + kk];
          stageA[m * APITCH + kk]     = (__bf16)v.x;
          stageA[m * APITCH + kk + 1] = (__bf16)v.y;
        }
      } else {                                // from recurrent state (bf16)
        for (int i = tid; i < BB * KCHUNK; i += TPB) {
          int kk = i & (KCHUNK - 1);
          int m  = i >> 8;
          stageA[m * APITCH + kk] = hb[m * UU + (kbase - DD) + kk];
        }
      }
      __syncthreads();

      for (int k0 = 0; k0 < KCHUNK; k0 += 32) {
        // B fragment: K runs [8*hi, 8*hi+8) and [16+8*hi, 24+8*hi) of this wave's weight column (L2-hot)
        const __bf16* bp = &WgT[(size_t)(ncolA + nl) * CC + kbase + k0 + 8 * hi];
        __builtin_prefetch(bp + 32, 0, 1);    // global_prefetch_b8 of next k0
        v16bf bfrag = make_frag(((const v8bf*)bp)[0], ((const v8bf*)bp)[2]);
        #pragma unroll
        for (int mt = 0; mt < 4; ++mt) {
          const __bf16* ap = &stageA[(mt * 16 + nl) * APITCH + k0 + 8 * hi];
          v16bf afrag = make_frag(*(const v8bf*)ap, *(const v8bf*)(ap + 16));
          acc[mt] = __builtin_amdgcn_wmma_f32_16x16x32_bf16(
              false, afrag, false, bfrag, (short)0, acc[mt], false, false);
        }
      }
    }

    // Epilogue: sigmoid gate, multiply by combined, store bf16 a2.
    {
      const int n = ncolA + nl;               // column is lane-invariant per acc
      const float bgn = bg[n];
      #pragma unroll
      for (int mt = 0; mt < 4; ++mt) {
        #pragma unroll
        for (int r = 0; r < 8; ++r) {
          int m = mt * 16 + r + 8 * hi;
          float g = acc[mt][r] + bgn;
          g = 1.0f / (1.0f + __expf(-g));
          float cv = (n < DD) ? x[((size_t)m * TT + t) * DD + n]
                              : (float)hb[m * UU + (n - DD)];
          a2[(size_t)m * CC + n] = (__bf16)(cv * g);
        }
      }
    }
    gbar(cnt, gen);   // a2 complete

    // =================== Phase B: z GEMM + SiLU + LN ==================
    if (blk < 4) {
      const int mt = blk;                     // rows mt*16 .. mt*16+15
      const int nbase = wave * 64;            // 4 N-tiles per wave
      v8f accz[4] = {};

      for (int kc = 0; kc < CC / KCHUNK; ++kc) {
        const int kbase = kc * KCHUNK;
        __syncthreads();
        // Stage 16 rows x 256 K of a2 (8 KB) as uint4 copies.
        for (int i = tid; i < 16 * (KCHUNK / 8); i += TPB) {  // 32 uint4/row
          int row = i >> 5;
          int c16 = i & 31;
          *(uint4*)&stageB[row * APITCH + c16 * 8] =
              *(const uint4*)&a2[(size_t)(mt * 16 + row) * CC + kbase + c16 * 8];
        }
        __syncthreads();

        for (int k0 = 0; k0 < KCHUNK; k0 += 32) {
          const __bf16* ap = &stageB[nl * APITCH + k0 + 8 * hi];
          v16bf afrag = make_frag(*(const v8bf*)ap, *(const v8bf*)(ap + 16));
          #pragma unroll
          for (int j = 0; j < 4; ++j) {
            const __bf16* bp =
                &WfT[(size_t)(nbase + j * 16 + nl) * CC + kbase + k0 + 8 * hi];
            v16bf bfrag = make_frag(((const v8bf*)bp)[0], ((const v8bf*)bp)[2]);
            accz[j] = __builtin_amdgcn_wmma_f32_16x16x32_bf16(
                false, afrag, false, bfrag, (short)0, accz[j], false, false);
          }
        }
      }
      __syncthreads();

      // SiLU then park in LDS for the row reduction.
      #pragma unroll
      for (int j = 0; j < 4; ++j) {
        int n = nbase + j * 16 + nl;
        float bn = bfv[n];
        #pragma unroll
        for (int r = 0; r < 8; ++r) {
          int ml = r + 8 * hi;
          float z = accz[j][r] + bn;
          z = z / (1.0f + __expf(-z));        // silu
          zbuf[ml * UU + n] = z;
        }
      }
      __syncthreads();

      // LayerNorm: wave w owns rows 2w, 2w+1 (wave32 shuffle reduction).
      #pragma unroll
      for (int rr = 0; rr < 2; ++rr) {
        int ml = wave * 2 + rr;
        float s = 0.f, s2 = 0.f;
        #pragma unroll
        for (int j = 0; j < UU / 32; ++j) {
          float v = zbuf[ml * UU + lane + j * 32];
          s += v; s2 += v * v;
        }
        #pragma unroll
        for (int off = 16; off > 0; off >>= 1) {
          s  += __shfl_xor(s,  off, 32);
          s2 += __shfl_xor(s2, off, 32);
        }
        float mu  = s * (1.0f / UU);
        float var = s2 * (1.0f / UU) - mu * mu;
        float sc  = rsqrtf(var + EPSLN);
        int bglob = mt * 16 + ml;
        #pragma unroll
        for (int j = 0; j < UU / 32; ++j) {
          int n = lane + j * 32;
          float v = (zbuf[ml * UU + n] - mu) * sc * gamma[n] + beta[n];
          out[((size_t)bglob * TT + t) * UU + n] = v;   // y_t (f32)
          hb[bglob * UU + n] = (__bf16)v;               // h for step t+1
        }
      }
    }
    gbar(cnt, gen);   // h complete
  }
}

// ---------------------------------------------------------------------------
extern "C" void kernel_launch(void* const* d_in, const int* in_sizes, int n_in,
                              void* d_out, int out_size, void* d_ws, size_t ws_size,
                              hipStream_t stream) {
  (void)in_sizes; (void)n_in; (void)out_size; (void)ws_size;
  const float* x     = (const float*)d_in[0];
  const float* h0    = (const float*)d_in[1];
  const float* Wg    = (const float*)d_in[2];
  const float* bg    = (const float*)d_in[3];
  const float* Wf    = (const float*)d_in[4];
  const float* bfv   = (const float*)d_in[5];
  const float* gamma = (const float*)d_in[6];
  const float* beta  = (const float*)d_in[7];
  float* out = (float*)d_out;

  unsigned char* ws = (unsigned char*)d_ws;
  __bf16* WgT = (__bf16*)ws;                                   // 2 MB
  __bf16* WfT = (__bf16*)(ws + (size_t)2 * 1024 * 1024);       // 1 MB
  __bf16* hb  = (__bf16*)(ws + (size_t)3 * 1024 * 1024);       // 64 KB
  __bf16* a2  = (__bf16*)(ws + (size_t)3 * 1024 * 1024 + 64 * 1024);   // 128 KB
  unsigned* ctrl = (unsigned*)(ws + (size_t)3 * 1024 * 1024 + 192 * 1024);

  fgru_prep_weights<<<1024, 256, 0, stream>>>(Wg, Wf, WgT, WfT);
  fgru_prep_state<<<(BB * UU + 255) / 256, 256, 0, stream>>>(h0, hb, ctrl);
  fgru_persistent<<<NBLK, TPB, 0, stream>>>(x, bg, bfv, gamma, beta,
                                            WgT, WfT, hb, a2, ctrl, out);
}
